// Voxel_Merging_6373731467415
// MI455X (gfx1250) — compile-verified, matrix operand order checked by reading, not run on hardware
//
#include <hip/hip_runtime.h>
#include <cstdint>

typedef __attribute__((ext_vector_type(4)))  float        v4f;
typedef __attribute__((ext_vector_type(8)))  float        v8f;
typedef __attribute__((ext_vector_type(4)))  unsigned int v4u;
typedef __attribute__((ext_vector_type(8)))  unsigned int v8u;
typedef __attribute__((ext_vector_type(16))) __bf16       v16bf;

// Problem constants
constexpr int CC = 96, DD = 64, HH = 64, WW = 64;
constexpr int OC = 192, K8C = 768;
constexpr int NPOS = 2 * 32 * 32 * 32;          // 65536 merged positions
constexpr int ROWS = 64, THREADS = 512;         // 64 rows / 16 waves per block
constexpr int KC = 32, NCHUNK = K8C / KC;       // 24 K-chunks of 32
constexpr int ASTRIDE = 388;                    // u32 per A row (384 data + 4 pad)
constexpr int WSTRIDE = 20;                     // u32 per W col per chunk (16 + 4 pad)
constexpr float EPS = 1e-5f, INV_N = 1.0f / 768.0f;

// LDS layout (u32 units)
constexpr int OFF_AHI = 0;
constexpr int OFF_ALO = OFF_AHI + ROWS * ASTRIDE;   // 24832
constexpr int OFF_W   = OFF_ALO + ROWS * ASTRIDE;   // 49664
constexpr int WBUFSZ  = OC * WSTRIDE;               // 3840 (one plane, one buffer)
constexpr int OFF_MU  = OFF_W + 2 * 2 * WBUFSZ;     // 65024 (2 buffers x hi/lo)
constexpr int OFF_RR  = OFF_MU + ROWS;              // 65088
constexpr int LDS_U32 = OFF_RR + ROWS;              // 65152 u32 = 260608 B

__device__ __forceinline__ uint32_t bfr(float f) {          // f32 -> bf16 bits (RNE)
    uint32_t u = __float_as_uint(f);
    return (u + 0x7FFFu + ((u >> 16) & 1u)) >> 16;
}
__device__ __forceinline__ float bff(uint32_t h) { return __uint_as_float(h << 16); }

struct HL { uint32_t hi, lo; };
__device__ __forceinline__ HL split2(float a, float b) {    // two f32 -> packed bf16 hi/lo
    const uint32_t ha = bfr(a), hb = bfr(b);
    const float ra = a - bff(ha), rb = b - bff(hb);
    HL r;
    r.hi = ha | (hb << 16);
    r.lo = bfr(ra) | (bfr(rb) << 16);
    return r;
}

// Prep: G[o] = sum_c gamma[c]*W[c][o]; Bt[o] = sum_c beta[c]*W[c][o]  (for LN epilogue)
__global__ void prep_gbt(const float* __restrict__ Wm, const float* __restrict__ gamma,
                         const float* __restrict__ beta, float* __restrict__ gbt) {
    const int o = threadIdx.x;
    float g = 0.0f, bt = 0.0f;
    for (int c = 0; c < K8C; ++c) {
        const float w = Wm[(long)c * OC + o];
        g  += gamma[c] * w;
        bt += beta[c]  * w;
    }
    gbt[o] = g;
    gbt[OC + o] = bt;
}

// Fused: merge + mask + stats + split-bf16 pack (phase 1); split-precision bf16 WMMA
// GEMM with LN applied in the epilogue (phase 2). One block = 64 rows x 192 cols.
__global__ __launch_bounds__(THREADS)
void voxel_merge_ln_gemm(const float* __restrict__ x,
                         const float* __restrict__ mask,
                         const float* __restrict__ gamma,
                         const float* __restrict__ Wm,      // [768, 192]
                         const float* __restrict__ gbt,     // [G(192), Bt(192)]
                         float* __restrict__ out,           // [NPOS, 192]
                         float* __restrict__ mout)          // [NPOS]
{
    extern __shared__ uint32_t sm[];
    const int t   = threadIdx.x;
    const int bid = blockIdx.x;                 // (b*32 + dz)*16 + hyb
    const int b   = bid >> 9;
    const int dz  = (bid >> 4) & 31;
    const int hyb = bid & 15;

    // ---------------- Phase 1: merge+mask, stats, pack bf16 hi/lo to LDS ----------------
    {
        const int row = t >> 3, seg = t & 7;    // row 0..63, seg = i*4+j*2+k
        const int rh = row >> 5, wx = row & 31;
        const int zi = 2 * dz + ((seg >> 2) & 1);
        const int yj = 2 * (2 * hyb + rh) + ((seg >> 1) & 1);
        const int xk = 2 * wx + (seg & 1);
        const long vox = (((long)(b * DD + zi) * HH) + yj) * WW + xk;
        const float* xp = x + vox * CC;
        const float m = mask[vox];

        uint32_t* aH = sm + OFF_AHI + row * ASTRIDE + seg * 48;
        uint32_t* aL = sm + OFF_ALO + row * ASTRIDE + seg * 48;

        float s = 0.0f, s2 = 0.0f;
        #pragma unroll
        for (int c = 0; c < CC; c += 8) {
            v4f v0 = *(const v4f*)(xp + c) * m;
            v4f v1 = *(const v4f*)(xp + c + 4) * m;
            s  += v0.x + v0.y + v0.z + v0.w + v1.x + v1.y + v1.z + v1.w;
            s2 += v0.x*v0.x + v0.y*v0.y + v0.z*v0.z + v0.w*v0.w
                + v1.x*v1.x + v1.y*v1.y + v1.z*v1.z + v1.w*v1.w;
            const HL p0 = split2(v0.x, v0.y);
            const HL p1 = split2(v0.z, v0.w);
            const HL p2 = split2(v1.x, v1.y);
            const HL p3 = split2(v1.z, v1.w);
            v4u hi, lo;
            hi.x = p0.hi; hi.y = p1.hi; hi.z = p2.hi; hi.w = p3.hi;
            lo.x = p0.lo; lo.y = p1.lo; lo.z = p2.lo; lo.w = p3.lo;
            *(v4u*)(aH + c / 2) = hi;
            *(v4u*)(aL + c / 2) = lo;
        }

        float ms = m;
        #pragma unroll
        for (int off = 1; off < 8; off <<= 1) {
            s  += __shfl_xor(s,  off, 32);
            s2 += __shfl_xor(s2, off, 32);
            ms += __shfl_xor(ms, off, 32);
        }
        const float mu  = s * INV_N;
        const float var = s2 * INV_N - mu * mu;
        const float rr  = __frsqrt_rn(var + EPS);
        if (seg == 0) {
            sm[OFF_MU + row] = __float_as_uint(mu);
            sm[OFF_RR + row] = __float_as_uint(rr);
            mout[bid * ROWS + row] = (ms > 0.0f) ? 1.0f : 0.0f;
        }
    }

    // ---------------- W staging machinery (gamma-scaled, split bf16, double-buffered) ----
    int colA[6], kk2A[6];
    #pragma unroll
    for (int it = 0; it < 6; ++it) {
        const int p = t + THREADS * it;         // 3072 k-pairs per chunk
        colA[it] = p % OC;
        kk2A[it] = p / OC;                      // 0..15
    }
    float wreg[12], greg[12];
    auto loadW = [&](int kb) {
        #pragma unroll
        for (int it = 0; it < 6; ++it) {
            const int k0 = kb + 2 * kk2A[it];
            wreg[2*it]   = Wm[(long)k0 * OC + colA[it]];
            wreg[2*it+1] = Wm[(long)(k0 + 1) * OC + colA[it]];
            greg[2*it]   = gamma[k0];
            greg[2*it+1] = gamma[k0 + 1];
        }
    };
    auto storeW = [&](int buf) {
        uint32_t* wh = sm + OFF_W + buf * (2 * WBUFSZ);
        uint32_t* wl = wh + WBUFSZ;
        #pragma unroll
        for (int it = 0; it < 6; ++it) {
            const HL p = split2(wreg[2*it] * greg[2*it], wreg[2*it+1] * greg[2*it+1]);
            wh[colA[it] * WSTRIDE + kk2A[it]] = p.hi;
            wl[colA[it] * WSTRIDE + kk2A[it]] = p.lo;
        }
    };

    loadW(0);
    storeW(0);
    __syncthreads();

    // ---------------- Phase 2: split-precision bf16 WMMA GEMM ----------------
    const int wv = t >> 5, lane = t & 31;
    const int rt = wv >> 2;                     // row tile 0..3
    const int ct0 = (wv & 3) * 3;               // 3 col tiles per wave
    const int hh = lane >> 4, l16 = lane & 15;
    // A fragment K-pattern per 16-bit WMMA layout: lanes 0-15 -> K 0..7,16..23 of chunk
    const uint32_t* arow = sm + OFF_AHI + (rt * 16 + l16) * ASTRIDE + hh * 4;

    v8f acc[3] = {};
    for (int ch = 0; ch < NCHUNK; ++ch) {
        if (ch + 1 < NCHUNK) loadW((ch + 1) * KC);   // global loads overlap WMMA below
        const int buf = ch & 1;

        const uint32_t* ap = arow + ch * 16;
        v8u ahu, alu;
        ahu.lo = *(const v4u*)(ap);
        ahu.hi = *(const v4u*)(ap + 8);
        alu.lo = *(const v4u*)(ap + ROWS * ASTRIDE);
        alu.hi = *(const v4u*)(ap + ROWS * ASTRIDE + 8);
        const v16bf aHi = __builtin_bit_cast(v16bf, ahu);
        const v16bf aLo = __builtin_bit_cast(v16bf, alu);

        const uint32_t* wb = sm + OFF_W + buf * (2 * WBUFSZ) + hh * 4;
        #pragma unroll
        for (int c3 = 0; c3 < 3; ++c3) {
            const int col = (ct0 + c3) * 16 + l16;
            const uint32_t* bp = wb + col * WSTRIDE;
            v8u bhu, blu;
            bhu.lo = *(const v4u*)(bp);
            bhu.hi = *(const v4u*)(bp + 8);
            blu.lo = *(const v4u*)(bp + WBUFSZ);
            blu.hi = *(const v4u*)(bp + WBUFSZ + 8);
            const v16bf bHi = __builtin_bit_cast(v16bf, bhu);
            const v16bf bLo = __builtin_bit_cast(v16bf, blu);
            // D = Ahi*Bhi + Ahi*Blo + Alo*Bhi + C   (~fp32 accuracy, 3/8 the fp32-WMMA cost)
            acc[c3] = __builtin_amdgcn_wmma_f32_16x16x32_bf16(
                false, aHi, false, bHi, (short)0, acc[c3], false, false);
            acc[c3] = __builtin_amdgcn_wmma_f32_16x16x32_bf16(
                false, aHi, false, bLo, (short)0, acc[c3], false, false);
            acc[c3] = __builtin_amdgcn_wmma_f32_16x16x32_bf16(
                false, aLo, false, bHi, (short)0, acc[c3], false, false);
        }

        if (ch + 1 < NCHUNK) storeW(buf ^ 1);
        __syncthreads();
    }

    // ---------------- Epilogue: apply LayerNorm affine per row/col ----------------
    const float* Gv = gbt;
    const float* Bv = gbt + OC;
    float murr[8], rrv[8];
    #pragma unroll
    for (int i = 0; i < 8; ++i) {
        const int rl = rt * 16 + hh * 8 + i;
        const float muv = __uint_as_float(sm[OFF_MU + rl]);
        const float rv  = __uint_as_float(sm[OFF_RR + rl]);
        rrv[i] = rv;
        murr[i] = muv * rv;
    }
    const long rbase = (long)bid * ROWS + rt * 16 + hh * 8;
    #pragma unroll
    for (int c3 = 0; c3 < 3; ++c3) {
        const int col = (ct0 + c3) * 16 + l16;
        const float gc = Gv[col], bc = Bv[col];
        #pragma unroll
        for (int i = 0; i < 8; ++i) {
            out[(rbase + i) * OC + col] = rrv[i] * acc[c3][i] - murr[i] * gc + bc;
        }
    }
}

extern "C" void kernel_launch(void* const* d_in, const int* in_sizes, int n_in,
                              void* d_out, int out_size, void* d_ws, size_t ws_size,
                              hipStream_t stream) {
    (void)in_sizes; (void)n_in; (void)out_size; (void)ws_size;

    const float* x     = (const float*)d_in[0];
    const float* mask  = (const float*)d_in[1];
    const float* gamma = (const float*)d_in[2];
    const float* beta  = (const float*)d_in[3];
    const float* Wm    = (const float*)d_in[4];

    float* out  = (float*)d_out;
    float* mout = out + (long)NPOS * OC;
    float* gbt  = (float*)d_ws;                  // 384 floats

    prep_gbt<<<dim3(1), dim3(OC), 0, stream>>>(Wm, gamma, beta, gbt);

    const int nblocks = 2 * 32 * 16;             // 1024 blocks of 512 threads
    const size_t ldsBytes = (size_t)LDS_U32 * sizeof(uint32_t);  // 260608 B
    voxel_merge_ln_gemm<<<dim3(nblocks), dim3(THREADS), ldsBytes, stream>>>(
        x, mask, gamma, Wm, gbt, out, mout);
}
